// SudokuRecurrentRelationalNet_86818468921628
// MI455X (gfx1250) — matrix operand order repaired and checked by reading, not compiled, not run on hardware
//
#include <hip/hip_runtime.h>

#define H   96
#define DE  16
#define NNODES 41472
#define NEDGES 829440
#define WPB 4            // waves per block

typedef __bf16 bf16_t;
typedef __attribute__((ext_vector_type(16))) __bf16 v16bf;
typedef __attribute__((ext_vector_type(8)))  __bf16 v8bf;
typedef __attribute__((ext_vector_type(8)))  float  v8f;

static __device__ __forceinline__ v8f wmma_bf16(v16bf a, v16bf b, v8f c) {
  // D(16x16 f32) = A(16x32 bf16) * B(32x16 bf16) + C
  return __builtin_amdgcn_wmma_f32_16x16x32_bf16(false, a, false, b, (short)0, c, false, false);
}

static __device__ __forceinline__ float sigmoidf_(float x) {
  return 1.0f / (1.0f + __expf(-x));
}

// build a 16x32 bf16 A fragment for this lane from an LDS row (stride SA bf16 elems).
// ISA layout: lane (m = lane&15, hi = lane>>4) holds K = kb*32 + hi*8 + {0..7} (elems 0-7)
//             and K = kb*32 + 16 + hi*8 + {0..7} (elems 8-15): two contiguous 16B chunks.
static __device__ __forceinline__ v16bf load_afrag(const bf16_t* row) {
  v8bf alo = *(const v8bf*)(row);
  v8bf ahi = *(const v8bf*)(row + 16);
  v16bf a;
  #pragma unroll
  for (int e = 0; e < 8; ++e) { a[e] = alo[e]; a[e + 8] = ahi[e]; }
  return a;
}

// convert 8 consecutive fp32 (two float4) to a 16B bf16 chunk
static __device__ __forceinline__ v8bf cvt8(float4 f0, float4 f1) {
  v8bf r;
  r[0] = (bf16_t)f0.x; r[1] = (bf16_t)f0.y; r[2] = (bf16_t)f0.z; r[3] = (bf16_t)f0.w;
  r[4] = (bf16_t)f1.x; r[5] = (bf16_t)f1.y; r[6] = (bf16_t)f1.z; r[7] = (bf16_t)f1.w;
  return r;
}

static __device__ __forceinline__ v8bf zero8() {
  v8bf z;
  #pragma unroll
  for (int e = 0; e < 8; ++e) z[e] = (bf16_t)0.0f;
  return z;
}

// ---- one MLP layer on a 16-row tile: out[16,96] = act( in[16,32*KSTEPS] @ W + b ) ----
// A fragments are kb-invariant across n-tiles: preload once into registers.
// Wp is bf16 packed fragment-major: Wp[(kb*96 + n)*32 + t], slot t holds K = 8*(t>>4) + (t&7) + 2*(t&8) within block.
// C/D layout: vgpr r holds (M = r + 8*hi, N = (lane&15) + 16*nt).
template<int KSTEPS, int SA, bool RELU>
static __device__ __forceinline__ void mlp_layer_bf(
    const bf16_t* __restrict__ inA, const bf16_t* __restrict__ Wp,
    const float* __restrict__ bias, bf16_t* __restrict__ outA, int lane)
{
  const int m  = lane & 15;
  const int hi = lane >> 4;
  v16bf afr[KSTEPS];
  #pragma unroll
  for (int kb = 0; kb < KSTEPS; ++kb)
    afr[kb] = load_afrag(inA + m * SA + kb * 32 + hi * 8);
  #pragma unroll 1
  for (int nt = 0; nt < 6; ++nt) {
    const int n = (lane & 15) + nt * 16;
    const float bv = bias[n];
    v8f c = {bv, bv, bv, bv, bv, bv, bv, bv};
    const bf16_t* bp = Wp + (((long)n) << 5) + hi * 16;   // kb steps = +96*32 elems
    #pragma unroll
    for (int kb = 0; kb < KSTEPS; ++kb)
      c = wmma_bf16(afr[kb], *(const v16bf*)(bp + ((long)kb * 96 << 5)), c);
    #pragma unroll
    for (int r = 0; r < 8; ++r) {
      float v = c[r];
      if (RELU) v = fmaxf(v, 0.0f);
      outA[(r + hi * 8) * 104 + n] = (bf16_t)v;
    }
  }
}

// ---- final edge layer: compute tile and atomically scatter-add rows (fp32) into agg[src] ----
template<int KSTEPS, int SA>
static __device__ __forceinline__ void mlp_layer_scatter_bf(
    const bf16_t* __restrict__ inA, const bf16_t* __restrict__ Wp,
    const float* __restrict__ bias, float* __restrict__ agg,
    const int* __restrict__ sidx, int lane)
{
  const int m  = lane & 15;
  const int hi = lane >> 4;
  v16bf afr[KSTEPS];
  #pragma unroll
  for (int kb = 0; kb < KSTEPS; ++kb)
    afr[kb] = load_afrag(inA + m * SA + kb * 32 + hi * 8);
  #pragma unroll 1
  for (int nt = 0; nt < 6; ++nt) {
    const int n = (lane & 15) + nt * 16;
    const float bv = bias[n];
    v8f c = {bv, bv, bv, bv, bv, bv, bv, bv};
    const bf16_t* bp = Wp + (((long)n) << 5) + hi * 16;
    #pragma unroll
    for (int kb = 0; kb < KSTEPS; ++kb)
      c = wmma_bf16(afr[kb], *(const v16bf*)(bp + ((long)kb * 96 << 5)), c);
    #pragma unroll
    for (int r = 0; r < 8; ++r) {
      const int mm = r + hi * 8;
      __hip_atomic_fetch_add(&agg[(long)sidx[mm] * H + n], (float)c[r],
                             __ATOMIC_RELAXED, __HIP_MEMORY_SCOPE_AGENT);
    }
  }
}

// ===================== prep kernels =====================
__global__ void zero_kernel(float* __restrict__ p, long n) {
  long i = (long)blockIdx.x * blockDim.x + threadIdx.x;
  if (i < n) p[i] = 0.0f;
}

__global__ void cvt_bf16_kernel(const float* __restrict__ src, bf16_t* __restrict__ dst, long n) {
  long i = (long)blockIdx.x * blockDim.x + threadIdx.x;
  if (i < n) dst[i] = (bf16_t)src[i];
}

// pack W[K][Nout] (fp32) -> Wp[(kb*Nout + n)*32 + t] (bf16), zero-padded past K
__global__ void pack_weight_kernel(const float* __restrict__ W, bf16_t* __restrict__ Wp,
                                   int K, int KB, int Nout) {
  long i = (long)blockIdx.x * blockDim.x + threadIdx.x;
  long total = (long)KB * Nout * 32;
  if (i >= total) return;
  int  t  = (int)(i & 31);
  long cc = i >> 5;                 // kb*Nout + n
  int  n  = (int)(cc % Nout);
  int  kb = (int)(cc / Nout);
  int  k  = kb * 32 + ((t >> 4) << 3) + (t & 7) + ((t & 8) << 1);
  Wp[i] = (k < K) ? (bf16_t)W[(long)k * Nout + n] : (bf16_t)0.0f;
}

// ===================== kernel 1: edge message MLP + segment-sum =====================
#define EK_SA 232                                 // row stride (bf16), 16B-aligned, conflict-free
#define EK_WAVE_ELEMS (16*EK_SA + 16*104 + 16*104 + 32)   // inA | actA | actB | sidx(16 int)

__global__ void edge_msg_kernel(
    const bf16_t* __restrict__ nodesbf, const float* __restrict__ edgef,
    const long long* __restrict__ edges,
    const bf16_t* __restrict__ W1p, const float* __restrict__ b1,
    const bf16_t* __restrict__ W2p, const float* __restrict__ b2,
    const bf16_t* __restrict__ W3p, const float* __restrict__ b3,
    float* __restrict__ agg)
{
  extern __shared__ bf16_t smem[];
  const int wave = threadIdx.x >> 5;
  const int lane = threadIdx.x & 31;
  bf16_t* inA  = smem + wave * EK_WAVE_ELEMS;   // 16 x 232 : [src(96) | dst(96) | ef(16) | pad(16)]
  bf16_t* actA = inA + 16 * EK_SA;              // 16 x 104
  bf16_t* actB = actA + 16 * 104;               // 16 x 104
  int*    sidx = (int*)(actB + 16 * 104);       // 16 src node ids

  const int tile = blockIdx.x * WPB + wave;     // grid divides exactly
  const int e0   = tile * 16;

  // stage rows: 2 lanes per row, pure 16B bf16 copies for the node gather
  {
    const int r = lane >> 1;
    const int half = lane & 1;
    const long long s = edges[(long)(e0 + r) * 2 + 0];
    const long long d = edges[(long)(e0 + r) * 2 + 1];
    if (half == 0) sidx[r] = (int)s;
    const uint4* ps = (const uint4*)(nodesbf + s * H);   // 12 x 16B
    const uint4* pd = (const uint4*)(nodesbf + d * H);
    uint4* row = (uint4*)(inA + r * EK_SA);
    #pragma unroll
    for (int i = half; i < 12; i += 2) row[i] = ps[i];
    #pragma unroll
    for (int i = half; i < 12; i += 2) row[12 + i] = pd[i];
    // edge features: 16 fp32 -> 16 bf16 (one 16B chunk per half-lane)
    const float4* pe = (const float4*)(edgef + (long)(e0 + r) * DE);
    float4 f0 = pe[half * 2], f1 = pe[half * 2 + 1];
    *(v8bf*)(inA + r * EK_SA + 192 + half * 8) = cvt8(f0, f1);
    // zero-pad K = 208..223
    *(v8bf*)(inA + r * EK_SA + 208 + half * 8) = zero8();
  }
  __syncthreads();

  mlp_layer_bf<7, EK_SA, true>(inA, W1p, b1, actA, lane);   // 208(pad 224) -> 96, relu
  __syncthreads();
  mlp_layer_bf<3, 104, true>(actA, W2p, b2, actB, lane);    // 96 -> 96, relu
  __syncthreads();
  mlp_layer_scatter_bf<3, 104>(actB, W3p, b3, agg, sidx, lane); // 96 -> 96, fp32 atomic scatter
}

// ===================== kernel 2: node MLP + LSTM cell + output head =====================
#define NK_SA 136                                 // [agg|puzzle] row stride (bf16), pad 112->128
#define NK_BF_ELEMS (16*NK_SA + 16*104 + 16*104 + 16*104)  // inA | actA | actB | hA
#define NK_WAVE_ELEMS (NK_BF_ELEMS + 16*100*2)             // + hnewF (fp32, 16 x 100)

__global__ void node_update_kernel(
    const float* __restrict__ agg, const float* __restrict__ puzzle,
    const float* __restrict__ hin, const float* __restrict__ cin,
    const bf16_t* __restrict__ W1p, const float* __restrict__ b1,
    const bf16_t* __restrict__ W2p, const float* __restrict__ b2,
    const bf16_t* __restrict__ W3p, const float* __restrict__ b3,
    const bf16_t* __restrict__ WihP, const float* __restrict__ bih,
    const bf16_t* __restrict__ WhhP, const float* __restrict__ bhh,
    const float* __restrict__ Wout, const float* __restrict__ bout,
    float* __restrict__ out)
{
  extern __shared__ bf16_t smem[];
  const int wave = threadIdx.x >> 5;
  const int lane = threadIdx.x & 31;
  bf16_t* inA   = smem + wave * NK_WAVE_ELEMS;  // 16 x 136 : [agg(96) | puzzle(16) | pad(16)]
  bf16_t* actA  = inA + 16 * NK_SA;             // 16 x 104
  bf16_t* actB  = actA + 16 * 104;              // 16 x 104
  bf16_t* hA    = actB + 16 * 104;              // 16 x 104
  float*  hnewF = (float*)(hA + 16 * 104);      // 16 x 100 fp32

  const int tile = blockIdx.x * WPB + wave;
  const int v0   = tile * 16;

  // stage [agg | puzzle] rows (fp32 -> bf16), 2 lanes per row
  {
    const int r = lane >> 1;
    const int half = lane & 1;
    const float4* pa = (const float4*)(agg + (long)(v0 + r) * H);       // 24 float4
    bf16_t* row = inA + r * NK_SA;
    #pragma unroll
    for (int i = half; i < 12; i += 2)                                  // chunk i = floats 8i..8i+7
      *(v8bf*)(row + i * 8) = cvt8(pa[2 * i], pa[2 * i + 1]);
    const float4* pp = (const float4*)(puzzle + (long)(v0 + r) * DE);
    *(v8bf*)(row + 96 + half * 8) = cvt8(pp[half * 2], pp[half * 2 + 1]);
    *(v8bf*)(row + 112 + half * 8) = zero8();                           // pad K = 112..127
    // stage h tile (fp32 -> bf16)
    const float4* ph = (const float4*)(hin + (long)(v0 + r) * H);
    #pragma unroll
    for (int i = half; i < 12; i += 2)
      *(v8bf*)(hA + r * 104 + i * 8) = cvt8(ph[2 * i], ph[2 * i + 1]);
  }
  __syncthreads();
  mlp_layer_bf<4, NK_SA, true >(inA,  W1p, b1, actA, lane);   // 112(pad 128) -> 96, relu
  __syncthreads();
  mlp_layer_bf<3, 104, true >(actA, W2p, b2, actB, lane);     // 96 -> 96, relu
  __syncthreads();
  mlp_layer_bf<3, 104, false>(actB, W3p, b3, actA, lane);     // 96 -> 96 : x (bf16)
  __syncthreads();

  float* outH1 = out;
  float* outH2 = out + (long)NNODES * H;
  float* outC  = out + 2L * NNODES * H;
  float* outO  = out + 3L * NNODES * H;

  const int m  = lane & 15;
  const int hi = lane >> 4;

  // preload jt-invariant A fragments for x (actA) and h (hA): 6 x 8 VGPRs
  v16bf ax[3], ah[3];
  #pragma unroll
  for (int kb = 0; kb < 3; ++kb) {
    ax[kb] = load_afrag(actA + m * 104 + kb * 32 + hi * 8);
    ah[kb] = load_afrag(hA   + m * 104 + kb * 32 + hi * 8);
  }

  // gates: per 16-wide j-tile keep i/f/g/o accumulators live, sharing each A fragment.
  // Wih/Whh packed with Nout=384: frag at Wp[(kb*384 + col)*32 + hi*16], col = gate*96 + j.
  #pragma unroll 1
  for (int jt = 0; jt < 6; ++jt) {
    const int j = (lane & 15) + jt * 16;     // column within one gate (0..95)
    const float bi = bih[j]       + bhh[j];
    const float bf = bih[96 + j]  + bhh[96 + j];
    const float bg = bih[192 + j] + bhh[192 + j];
    const float bo = bih[288 + j] + bhh[288 + j];
    v8f ci = {bi,bi,bi,bi,bi,bi,bi,bi};
    v8f cf = {bf,bf,bf,bf,bf,bf,bf,bf};
    v8f cg = {bg,bg,bg,bg,bg,bg,bg,bg};
    v8f co = {bo,bo,bo,bo,bo,bo,bo,bo};
    #pragma unroll
    for (int kb = 0; kb < 3; ++kb) {         // x @ W_ih
      const bf16_t* bp = WihP + (((long)(kb * 384 + j)) << 5) + hi * 16;
      ci = wmma_bf16(ax[kb], *(const v16bf*)(bp),                 ci);
      cf = wmma_bf16(ax[kb], *(const v16bf*)(bp + (96L  << 5)),   cf);
      cg = wmma_bf16(ax[kb], *(const v16bf*)(bp + (192L << 5)),   cg);
      co = wmma_bf16(ax[kb], *(const v16bf*)(bp + (288L << 5)),   co);
    }
    #pragma unroll
    for (int kb = 0; kb < 3; ++kb) {         // h @ W_hh
      const bf16_t* bp = WhhP + (((long)(kb * 384 + j)) << 5) + hi * 16;
      ci = wmma_bf16(ah[kb], *(const v16bf*)(bp),                 ci);
      cf = wmma_bf16(ah[kb], *(const v16bf*)(bp + (96L  << 5)),   cf);
      cg = wmma_bf16(ah[kb], *(const v16bf*)(bp + (192L << 5)),   cg);
      co = wmma_bf16(ah[kb], *(const v16bf*)(bp + (288L << 5)),   co);
    }
    // LSTM elementwise (fp32): i,f,g,o for (row mm, col j) all live in this lane
    #pragma unroll
    for (int r = 0; r < 8; ++r) {
      const int  mm   = r + hi * 8;
      const long node = v0 + mm;
      const float iv = sigmoidf_(ci[r]);
      const float fv = sigmoidf_(cf[r]);
      const float gv = tanhf(cg[r]);
      const float ov = sigmoidf_(co[r]);
      const float cn = fv * cin[node * H + j] + iv * gv;
      const float hn = ov * tanhf(cn);
      outH1[node * H + j] = hn;
      outH2[node * H + j] = hn;
      outC [node * H + j] = cn;
      hnewF[mm * 100 + j] = hn;              // fp32 stage for the output head
    }
  }
  __syncthreads();

  // output head (fp32): out[16,10] = h_new[16,96] @ Wout[96,10] + bout (160 dots, 5/lane)
  #pragma unroll
  for (int q = 0; q < 5; ++q) {
    const int idx = lane * 5 + q;            // 0..159
    const int mm  = idx / 10;
    const int col = idx % 10;
    float s = bout[col];
    for (int k = 0; k < H; ++k) s += hnewF[mm * 100 + k] * Wout[k * 10 + col];
    outO[(long)(v0 + mm) * 10 + col] = s;
  }
}

extern "C" void kernel_launch(void* const* d_in, const int* in_sizes, int n_in,
                              void* d_out, int out_size, void* d_ws, size_t ws_size,
                              hipStream_t stream) {
  const float*     puzzle = (const float*)d_in[0];
  const float*     nodes  = (const float*)d_in[1];
  const float*     edgef  = (const float*)d_in[2];
  const float*     h      = (const float*)d_in[3];
  const float*     c      = (const float*)d_in[4];
  const long long* edges  = (const long long*)d_in[5];
  const float* msg_W1 = (const float*)d_in[6];
  const float* msg_b1 = (const float*)d_in[7];
  const float* msg_W2 = (const float*)d_in[8];
  const float* msg_b2 = (const float*)d_in[9];
  const float* msg_W3 = (const float*)d_in[10];
  const float* msg_b3 = (const float*)d_in[11];
  const float* post_W1 = (const float*)d_in[12];
  const float* post_b1 = (const float*)d_in[13];
  const float* post_W2 = (const float*)d_in[14];
  const float* post_b2 = (const float*)d_in[15];
  const float* post_W3 = (const float*)d_in[16];
  const float* post_b3 = (const float*)d_in[17];
  const float* W_ih = (const float*)d_in[18];
  const float* b_ih = (const float*)d_in[19];
  const float* W_hh = (const float*)d_in[20];
  const float* b_hh = (const float*)d_in[21];
  const float* out_W = (const float*)d_in[22];
  const float* out_b = (const float*)d_in[23];

  // ---- workspace carve-up (all chunks 32B-aligned) ----
  char* ws = (char*)d_ws;
  float*  agg     = (float*)ws;   ws += (long)NNODES * H * sizeof(float);   // fp32 accumulator
  bf16_t* nodesbf = (bf16_t*)ws;  ws += (long)NNODES * H * sizeof(bf16_t);  // bf16 node features
  bf16_t* W1p = (bf16_t*)ws;  ws += 7L * 96 * 32 * 2;    // msg_W1: K=208 pad 224
  bf16_t* W2p = (bf16_t*)ws;  ws += 3L * 96 * 32 * 2;    // msg_W2
  bf16_t* W3p = (bf16_t*)ws;  ws += 3L * 96 * 32 * 2;    // msg_W3
  bf16_t* P1p = (bf16_t*)ws;  ws += 4L * 96 * 32 * 2;    // post_W1: K=112 pad 128
  bf16_t* P2p = (bf16_t*)ws;  ws += 3L * 96 * 32 * 2;    // post_W2
  bf16_t* P3p = (bf16_t*)ws;  ws += 3L * 96 * 32 * 2;    // post_W3
  bf16_t* WihP = (bf16_t*)ws; ws += 3L * 384 * 32 * 2;   // W_ih
  bf16_t* WhhP = (bf16_t*)ws; ws += 3L * 384 * 32 * 2;   // W_hh

  float* out = (float*)d_out;

  const long aggN = (long)NNODES * H;             // 3,981,312 = 15552 * 256
  zero_kernel<<<(int)(aggN / 256), 256, 0, stream>>>(agg, aggN);
  cvt_bf16_kernel<<<(int)(aggN / 256), 256, 0, stream>>>(nodes, nodesbf, aggN);

  pack_weight_kernel<<<(7 * 96 * 32 + 255) / 256, 256, 0, stream>>>(msg_W1, W1p, 208, 7, 96);
  pack_weight_kernel<<<(3 * 96 * 32 + 255) / 256, 256, 0, stream>>>(msg_W2, W2p, 96, 3, 96);
  pack_weight_kernel<<<(3 * 96 * 32 + 255) / 256, 256, 0, stream>>>(msg_W3, W3p, 96, 3, 96);
  pack_weight_kernel<<<(4 * 96 * 32 + 255) / 256, 256, 0, stream>>>(post_W1, P1p, 112, 4, 96);
  pack_weight_kernel<<<(3 * 96 * 32 + 255) / 256, 256, 0, stream>>>(post_W2, P2p, 96, 3, 96);
  pack_weight_kernel<<<(3 * 96 * 32 + 255) / 256, 256, 0, stream>>>(post_W3, P3p, 96, 3, 96);
  pack_weight_kernel<<<(3 * 384 * 32 + 255) / 256, 256, 0, stream>>>(W_ih, WihP, 96, 3, 384);
  pack_weight_kernel<<<(3 * 384 * 32 + 255) / 256, 256, 0, stream>>>(W_hh, WhhP, 96, 3, 384);

  edge_msg_kernel<<<NEDGES / 16 / WPB, 32 * WPB,
                    WPB * EK_WAVE_ELEMS * sizeof(bf16_t), stream>>>(
      nodesbf, edgef, edges, W1p, msg_b1, W2p, msg_b2, W3p, msg_b3, agg);

  node_update_kernel<<<NNODES / 16 / WPB, 32 * WPB,
                       WPB * NK_WAVE_ELEMS * sizeof(bf16_t), stream>>>(
      agg, puzzle, h, c, P1p, post_b1, P2p, post_b2, P3p, post_b3,
      WihP, b_ih, WhhP, b_hh, out_W, out_b, out);
}
